// AttentionVae_3865470567038
// MI455X (gfx1250) — compile-verified, matrix operand order checked by reading, not run on hardware
//
#include <hip/hip_runtime.h>
#include <math.h>

// ---------------- model config (matches reference) ----------------
#define Ecfg   768
#define Hcfg   12
#define Dcfg   64
#define Tcfg   1024
#define Bcfg   2
#define Vcfg   32000
#define MROWS  (Bcfg * Tcfg)   // 2048

typedef __attribute__((ext_vector_type(16))) __bf16 v16bf;
typedef __attribute__((ext_vector_type(8)))  float  v8f;

#define WMMA_BF16(a, b, c) \
  __builtin_amdgcn_wmma_f32_16x16x32_bf16(false, (a), false, (b), (short)0, (c), false, false)

__device__ __forceinline__ v8f zero8() {
  v8f z;
#pragma unroll
  for (int i = 0; i < 8; ++i) z[i] = 0.0f;
  return z;
}

// ---- gfx1250 async global->LDS copy (ASYNCcnt-tracked, bypasses VGPRs) ----
// dst operand is a VGPR holding the LDS byte address (low 32 bits of the flat
// shared-aperture address, per ISA 10.2).
__device__ __forceinline__ void async_cp_b128(void* lds_ptr, const void* gaddr) {
  const unsigned lds_addr = (unsigned)(size_t)lds_ptr;
  asm volatile("global_load_async_to_lds_b128 %0, %1, off"
               :: "v"(lds_addr), "v"(gaddr)
               : "memory");
}
__device__ __forceinline__ void wait_async0() {
  asm volatile("s_wait_asynccnt 0" ::: "memory");
}

// A fragment (16x32 bf16, M x K). lanes 0-15: rows 0-15, K {0..7,16..23};
// lanes 16-31: rows 0-15, K {8..15,24..31}.
__device__ __forceinline__ v16bf frag_A(const __bf16* base, int ld, int row0, int k0, int lane) {
  const int r  = row0 + (lane & 15);
  const int kg = k0 + ((lane & 16) ? 8 : 0);
  v16bf f;
#pragma unroll
  for (int i = 0; i < 16; ++i) {
    const int k = kg + ((i < 8) ? i : (i + 8));
    f[i] = base[r * ld + k];
  }
  return f;
}

// B fragment (32x16 bf16, K x N), symmetric to A over the N dimension.
__device__ __forceinline__ v16bf frag_B(const __bf16* base, int ld, int col0, int k0, int lane) {
  const int n  = col0 + (lane & 15);
  const int kg = k0 + ((lane & 16) ? 8 : 0);
  v16bf f;
#pragma unroll
  for (int i = 0; i < 16; ++i) {
    const int k = kg + ((i < 8) ? i : (i + 8));
    f[i] = base[k * ld + n];
  }
  return f;
}

__device__ __forceinline__ float gelu_exact(float x) {
  return 0.5f * x * (1.0f + erff(x * 0.70710678118654752f));
}

// ======================= GEMM (bias + optional GELU + optional residual) ==================
// C[M,N] = act(A[M,K] @ W[K,N] + bias) + res ;  A,W,res fp32 row-major, bf16 WMMA core.
// Next K-tile is prefetched into double-buffered LDS via global_load_async_to_lds_b128
// while the current tile's WMMAs run; fp32->bf16 conversion happens LDS->LDS.
#define BM 128
#define BN 64
#define BK 32

__global__ __launch_bounds__(256)
void gemm_kernel(const float* __restrict__ A, const float* __restrict__ W,
                 const float* __restrict__ bias, const float* __restrict__ res,
                 float* __restrict__ C, int M, int N, int K, int act) {
  __shared__ float  Af[2][BM][BK];  // 2 x 16 KB fp32 stage (async target)
  __shared__ float  Bf[2][BK][BN];  // 2 x  8 KB fp32 stage (async target)
  __shared__ __bf16 As[BM][BK];     // 8 KB bf16 tile (WMMA fragment source)
  __shared__ __bf16 Bs[BK][BN];     // 4 KB

  const int tid  = threadIdx.x;
  const int lane = tid & 31;
  const int wave = tid >> 5;
  const int wm   = wave & 3;        // 4 wave-rows of 32
  const int wn   = wave >> 2;       // 2 wave-cols of 32
  const int m0   = blockIdx.y * BM;
  const int n0   = blockIdx.x * BN;

  v8f acc[2][2];
#pragma unroll
  for (int i = 0; i < 2; ++i)
#pragma unroll
    for (int j = 0; j < 2; ++j) acc[i][j] = zero8();

  const int nk = K / BK;

  // ---- issue async copies of K-tile `kt` into stage buffer `buf` ----
  auto issue_tile = [&](int kt, int buf) {
    const int k0 = kt * BK;
    // A tile: 128x32 fp32 = 1024 x 16B chunks, 4 per thread
#pragma unroll
    for (int i = 0; i < 4; ++i) {
      const int e = tid * 4 + i;
      const int r = e >> 3;          // 8 chunks per 32-float row
      const int c = (e & 7) * 4;
      async_cp_b128(&Af[buf][r][c], &A[(size_t)(m0 + r) * K + k0 + c]);
    }
    // B tile: 32x64 fp32 = 512 x 16B chunks, 2 per thread
#pragma unroll
    for (int i = 0; i < 2; ++i) {
      const int e = tid * 2 + i;
      const int r = e >> 4;          // 16 chunks per 64-float row
      const int c = (e & 15) * 4;
      async_cp_b128(&Bf[buf][r][c], &W[(size_t)(k0 + r) * N + n0 + c]);
    }
  };

  issue_tile(0, 0);

  for (int kt = 0; kt < nk; ++kt) {
    const int cur = kt & 1;
    wait_async0();        // our async writes to stage[cur] (and none newer) done
    __syncthreads();      // all waves' stage[cur] writes visible

    if (kt + 1 < nk) issue_tile(kt + 1, 1 - cur);  // overlap with convert+WMMA

    // LDS->LDS fp32 -> bf16 conversion (same chunk mapping as the async issue)
#pragma unroll
    for (int i = 0; i < 4; ++i) {
      const int e = tid * 4 + i;
      const int r = e >> 3;
      const int c = (e & 7) * 4;
      const float4 f = *(const float4*)&Af[cur][r][c];
      As[r][c + 0] = (__bf16)f.x;
      As[r][c + 1] = (__bf16)f.y;
      As[r][c + 2] = (__bf16)f.z;
      As[r][c + 3] = (__bf16)f.w;
    }
#pragma unroll
    for (int i = 0; i < 2; ++i) {
      const int e = tid * 2 + i;
      const int r = e >> 4;
      const int c = (e & 15) * 4;
      const float4 f = *(const float4*)&Bf[cur][r][c];
      Bs[r][c + 0] = (__bf16)f.x;
      Bs[r][c + 1] = (__bf16)f.y;
      Bs[r][c + 2] = (__bf16)f.z;
      Bs[r][c + 3] = (__bf16)f.w;
    }
    __syncthreads();

    const v16bf a0 = frag_A(&As[0][0], BK, wm * 32 + 0,  0, lane);
    const v16bf a1 = frag_A(&As[0][0], BK, wm * 32 + 16, 0, lane);
    const v16bf b0 = frag_B(&Bs[0][0], BN, wn * 32 + 0,  0, lane);
    const v16bf b1 = frag_B(&Bs[0][0], BN, wn * 32 + 16, 0, lane);
    acc[0][0] = WMMA_BF16(a0, b0, acc[0][0]);
    acc[0][1] = WMMA_BF16(a0, b1, acc[0][1]);
    acc[1][0] = WMMA_BF16(a1, b0, acc[1][0]);
    acc[1][1] = WMMA_BF16(a1, b1, acc[1][1]);
    __syncthreads();      // As/Bs fully consumed before next conversion pass
  }

  // epilogue: C layout -> vgpr r holds row (r | r+8 by lane half), lane&15 = col
  const int mo = (lane & 16) ? 8 : 0;
  const int nn = lane & 15;
#pragma unroll
  for (int i = 0; i < 2; ++i)
#pragma unroll
    for (int j = 0; j < 2; ++j) {
#pragma unroll
      for (int r = 0; r < 8; ++r) {
        const int gr = m0 + wm * 32 + i * 16 + r + mo;
        const int gc = n0 + wn * 32 + j * 16 + nn;
        float v = acc[i][j][r];
        if (bias) v += bias[gc];
        if (act)  v = gelu_exact(v);
        const size_t off = (size_t)gr * N + gc;
        if (res) v += res[off];
        C[off] = v;
      }
    }
}

// ======================= Flash attention (one wave / 16-query tile) =======================
// q,k,v,y are [B, T, E] fp32 with head slice at column h*D. kv length == T.
__global__ __launch_bounds__(32)
void attn_kernel(const float* __restrict__ q, const float* __restrict__ k,
                 const float* __restrict__ v, float* __restrict__ y,
                 int T_, int causal) {
  const int qt   = blockIdx.x;   // query tile (16 rows)
  const int h    = blockIdx.y;
  const int b    = blockIdx.z;
  const int lane = threadIdx.x;

  __shared__ __bf16 Qs[16][64];  // 2 KB
  __shared__ __bf16 KT[64][32];  // 4 KB  [d][key]
  __shared__ __bf16 Vs[32][64];  // 4 KB  [key][d]
  __shared__ __bf16 Ps[16][32];  // 1 KB

  const size_t bo  = (size_t)b * T_ * Ecfg;
  const int    hco = h * Dcfg;

  for (int e = lane; e < 16 * 64; e += 32) {
    const int r = e >> 6, c = e & 63;
    Qs[r][c] = (__bf16)q[bo + (size_t)(qt * 16 + r) * Ecfg + hco + c];
  }
  __syncthreads();
  const v16bf qa0 = frag_A(&Qs[0][0], 64, 0, 0,  lane);
  const v16bf qa1 = frag_A(&Qs[0][0], 64, 0, 32, lane);

  const int mo = (lane & 16) ? 8 : 0;
  const int nn = lane & 15;

  float rm[8], rl[8];
  v8f oacc[4];
#pragma unroll
  for (int r = 0; r < 8; ++r) { rm[r] = -INFINITY; rl[r] = 0.0f; }
#pragma unroll
  for (int j = 0; j < 4; ++j) oacc[j] = zero8();

  const int kt_max = causal ? (qt * 16 + 15) / 32 : (T_ / 32 - 1);
  for (int kt = 0; kt <= kt_max; ++kt) {
    // stage K (transposed) and V tiles, fp32 -> bf16
    for (int e = lane; e < 32 * 64; e += 32) {
      const int r = e >> 6, c = e & 63;
      const size_t goff = bo + (size_t)(kt * 32 + r) * Ecfg + hco + c;
      KT[c][r] = (__bf16)k[goff];
      Vs[r][c] = (__bf16)v[goff];
    }
    __syncthreads();

    // S = Q @ K^T  (two 16x16 score frags over 32 keys; D=64 -> 2 chained WMMAs)
    v8f sacc[2];
#pragma unroll
    for (int c = 0; c < 2; ++c) {
      sacc[c] = zero8();
      sacc[c] = WMMA_BF16(qa0, frag_B(&KT[0][0], 32, c * 16, 0,  lane), sacc[c]);
      sacc[c] = WMMA_BF16(qa1, frag_B(&KT[0][0], 32, c * 16, 32, lane), sacc[c]);
    }

    // scale + causal mask + online softmax
    float p[2][8], mcur[8];
#pragma unroll
    for (int r = 0; r < 8; ++r) mcur[r] = -INFINITY;
#pragma unroll
    for (int c = 0; c < 2; ++c)
#pragma unroll
      for (int r = 0; r < 8; ++r) {
        float sv = sacc[c][r] * 0.125f;  // 1/sqrt(64)
        if (causal && (kt * 32 + c * 16 + nn > qt * 16 + r + mo)) sv = -INFINITY;
        p[c][r] = sv;
        mcur[r] = fmaxf(mcur[r], sv);
      }
#pragma unroll
    for (int o = 1; o < 16; o <<= 1)
#pragma unroll
      for (int r = 0; r < 8; ++r) mcur[r] = fmaxf(mcur[r], __shfl_xor(mcur[r], o, 32));

    float scale[8];
#pragma unroll
    for (int r = 0; r < 8; ++r) {
      const float mn = fmaxf(rm[r], mcur[r]);
      scale[r] = __expf(rm[r] - mn);  // exp(-inf - finite) = 0 on first tile
      rm[r] = mn;
    }
    float rs[8];
#pragma unroll
    for (int r = 0; r < 8; ++r) rs[r] = 0.0f;
#pragma unroll
    for (int c = 0; c < 2; ++c)
#pragma unroll
      for (int r = 0; r < 8; ++r) {
        const float e_ = __expf(p[c][r] - rm[r]);
        p[c][r] = e_;
        rs[r] += e_;
      }
#pragma unroll
    for (int o = 1; o < 16; o <<= 1)
#pragma unroll
      for (int r = 0; r < 8; ++r) rs[r] += __shfl_xor(rs[r], o, 32);
#pragma unroll
    for (int r = 0; r < 8; ++r) rl[r] = rl[r] * scale[r] + rs[r];
#pragma unroll
    for (int j = 0; j < 4; ++j)
#pragma unroll
      for (int r = 0; r < 8; ++r) oacc[j][r] *= scale[r];

    // re-fragment P through LDS (C layout -> A layout), then O += P @ V
#pragma unroll
    for (int c = 0; c < 2; ++c)
#pragma unroll
      for (int r = 0; r < 8; ++r) Ps[r + mo][c * 16 + nn] = (__bf16)p[c][r];
    __syncthreads();
    const v16bf pa = frag_A(&Ps[0][0], 32, 0, 0, lane);
#pragma unroll
    for (int j = 0; j < 4; ++j)
      oacc[j] = WMMA_BF16(pa, frag_B(&Vs[0][0], 64, j * 16, 0, lane), oacc[j]);
    __syncthreads();
  }

  // normalize and write out (fp32)
#pragma unroll
  for (int j = 0; j < 4; ++j)
#pragma unroll
    for (int r = 0; r < 8; ++r) {
      const int m = r + mo;
      y[bo + (size_t)(qt * 16 + m) * Ecfg + hco + j * 16 + nn] = oacc[j][r] / rl[r];
    }
}

// ======================= elementwise / LN / embedding =======================
__global__ __launch_bounds__(256)
void layernorm_kernel(const float* __restrict__ x, const float* __restrict__ w,
                      const float* __restrict__ b, float* __restrict__ y, int E_) {
  __shared__ float red[512];
  const int row = blockIdx.x;
  const int tid = threadIdx.x;
  const float* xr = x + (size_t)row * E_;
  float s = 0.0f, ss = 0.0f;
  for (int e = tid; e < E_; e += blockDim.x) { const float v = xr[e]; s += v; ss += v * v; }
  red[tid] = s; red[256 + tid] = ss;
  __syncthreads();
  for (int st = 128; st > 0; st >>= 1) {
    if (tid < st) { red[tid] += red[tid + st]; red[256 + tid] += red[256 + tid + st]; }
    __syncthreads();
  }
  const float mean = red[0] * (1.0f / E_);
  const float var  = red[256] * (1.0f / E_) - mean * mean;
  const float inv  = rsqrtf(var + 1e-5f);
  float* yr = y + (size_t)row * E_;
  for (int e = tid; e < E_; e += blockDim.x) yr[e] = (xr[e] - mean) * inv * w[e] + b[e];
}

__global__ __launch_bounds__(256)
void embed_kernel(const int* __restrict__ idx, const float* __restrict__ tok,
                  const float* __restrict__ pos, float* __restrict__ y, int T_, int E_) {
  const int row = blockIdx.x;
  const int t   = row % T_;
  const int id  = idx[row];
  const float* tr = tok + (size_t)id * E_;
  const float* pr = pos + (size_t)t * E_;
  float* yr = y + (size_t)row * E_;
  for (int e = threadIdx.x; e < E_; e += blockDim.x) yr[e] = tr[e] + pr[e];
}

// h4 [rows,4E] -> zk[rows,E], zv[rows,E]; z = mu + eps*exp(0.5*logvar)
__global__ __launch_bounds__(256)
void reparam_kernel(const float* __restrict__ h4, const float* __restrict__ eps,
                    float* __restrict__ zk, float* __restrict__ zv, int E_) {
  const size_t i = blockIdx.x;
  const float* hr = h4 + i * 4 * E_;
  const float* er = eps + i * 2 * E_;
  for (int j = threadIdx.x; j < E_; j += blockDim.x) {
    zk[i * E_ + j] = hr[j]      + er[j]      * expf(0.5f * hr[2 * E_ + j]);
    zv[i * E_ + j] = hr[E_ + j] + er[E_ + j] * expf(0.5f * hr[3 * E_ + j]);
  }
}

// ======================= host-side orchestration =======================
static void gemm(hipStream_t s, const float* A, const float* W, const float* bias,
                 const float* res, float* C, int M, int N, int K, int act) {
  dim3 g(N / BN, M / BM);
  gemm_kernel<<<g, 256, 0, s>>>(A, W, bias, res, C, M, N, K, act);
}

static void ln(hipStream_t s, const float* x, const float* w, const float* b, float* y) {
  layernorm_kernel<<<MROWS, 256, 0, s>>>(x, w, b, y, Ecfg);
}

static void attn(hipStream_t s, const float* q, const float* k, const float* v,
                 float* y, int causal) {
  attn_kernel<<<dim3(Tcfg / 16, Hcfg, Bcfg), 32, 0, s>>>(q, k, v, y, Tcfg, causal);
}

// Transformer block. For cross=true, d_in[pb..] is the 12-tensor "post" dict and
// K/V come directly from zk/zv. Offsets: blk{Wq+4,Wk+6,Wv+8,Wo+10,Wfc+12,Wpr+14},
// post{Wq+4,Wo+6,Wfc+8,Wpr+10}.
static void run_block(hipStream_t s, void* const* d_in, int pb, int causal, int cross,
                      float** x, float** xn, float* hb, float* qb, float* kb,
                      float* vb, float* ab, float* mb,
                      const float* zk, const float* zv) {
  auto F = [&](int i) { return (const float*)d_in[i]; };
  ln(s, *x, F(pb + 0), F(pb + 1), hb);
  gemm(s, hb, F(pb + 4), F(pb + 5), nullptr, qb, MROWS, Ecfg, Ecfg, 0);
  const float *Kin, *Vin;
  if (cross) { Kin = zk; Vin = zv; }
  else {
    gemm(s, hb, F(pb + 6), F(pb + 7), nullptr, kb, MROWS, Ecfg, Ecfg, 0);
    gemm(s, hb, F(pb + 8), F(pb + 9), nullptr, vb, MROWS, Ecfg, Ecfg, 0);
    Kin = kb; Vin = vb;
  }
  attn(s, qb, Kin, Vin, ab, causal);
  const int io = cross ? pb + 6 : pb + 10;  // Wo
  gemm(s, ab, F(io), F(io + 1), *x, *xn, MROWS, Ecfg, Ecfg, 0);
  { float* t = *x; *x = *xn; *xn = t; }
  ln(s, *x, F(pb + 2), F(pb + 3), hb);
  gemm(s, hb, F(io + 2), F(io + 3), nullptr, mb, MROWS, 4 * Ecfg, Ecfg, 1);      // GELU
  gemm(s, mb, F(io + 4), F(io + 5), *x, *xn, MROWS, Ecfg, 4 * Ecfg, 0);
  { float* t = *x; *x = *xn; *xn = t; }
}

extern "C" void kernel_launch(void* const* d_in, const int* in_sizes, int n_in,
                              void* d_out, int out_size, void* d_ws, size_t ws_size,
                              hipStream_t stream) {
  (void)in_sizes; (void)n_in; (void)out_size; (void)ws_size;
  const int*   idx = (const int*)d_in[0];
  const float* eps = (const float*)d_in[1];
  auto F = [&](int i) { return (const float*)d_in[i]; };

  // input index map (flattened in setup_inputs() order):
  // 0 idx, 1 eps | enc: 2 tok, 3 pos, 4+16*l blocks, 68 lnfw, 69 lnfb, 70 head
  // dec: 71 tok, 72 pos, 73+16*l blocks, 137 lnfw, 138 lnfb, 139 post(12), 151 lnf2w, 152 lnf2b, 153 head

  float* ws = (float*)d_ws;
  const size_t S1 = (size_t)MROWS * Ecfg;       // 1.57M floats
  float* x0 = ws + 0 * S1;
  float* x1 = ws + 1 * S1;
  float* hb = ws + 2 * S1;
  float* qb = ws + 3 * S1;
  float* kb = ws + 4 * S1;
  float* vb = ws + 5 * S1;
  float* ab = ws + 6 * S1;
  float* zk = ws + 7 * S1;
  float* zv = ws + 8 * S1;
  float* mb = ws + 9 * S1;                      // [MROWS, 4E]

  float* x  = x0;
  float* xn = x1;

  // ---------------- encoder ----------------
  embed_kernel<<<MROWS, 256, 0, stream>>>(idx, F(2), F(3), x, Tcfg, Ecfg);
  for (int l = 0; l < 4; ++l)
    run_block(stream, d_in, 4 + 16 * l, /*causal=*/0, /*cross=*/0,
              &x, &xn, hb, qb, kb, vb, ab, mb, nullptr, nullptr);
  ln(stream, x, F(68), F(69), hb);
  gemm(stream, hb, F(70), nullptr, nullptr, mb, MROWS, 4 * Ecfg, Ecfg, 0);  // h4
  reparam_kernel<<<MROWS, 256, 0, stream>>>(mb, eps, zk, zv, Ecfg);

  // ---------------- decoder ----------------
  embed_kernel<<<MROWS, 256, 0, stream>>>(idx, F(71), F(72), x, Tcfg, Ecfg);
  for (int l = 0; l < 4; ++l)
    run_block(stream, d_in, 73 + 16 * l, /*causal=*/1, /*cross=*/0,
              &x, &xn, hb, qb, kb, vb, ab, mb, nullptr, nullptr);
  ln(stream, x, F(137), F(138), xn);
  { float* t = x; x = xn; xn = t; }
  run_block(stream, d_in, 139, /*causal=*/0, /*cross=*/1,
            &x, &xn, hb, qb, kb, vb, ab, mb, zk, zv);
  ln(stream, x, F(151), F(152), hb);
  gemm(stream, hb, F(153), nullptr, nullptr, (float*)d_out, MROWS, Vcfg, Ecfg, 0);
}
